// HATAFormerScaledDotProductAttention_10496900071839
// MI455X (gfx1250) — compile-verified, matrix operand order checked by reading, not run on hardware
//
#include <hip/hip_runtime.h>

// ---------------------------------------------------------------------------
// Fused SDPA, B=4,H=16,S=1024,D=64 (fp32 in/out), outputs (O, attn_weights).
//
// Pipeline (ws >= ~25.2 MB):
//   k1: Q*0.125,K fp32 -> bf16 copies in ws   (scale folded, exact: 2^-3)
//   k2: V fp32 -> bf16 TRANSPOSED (Vt[d][s])  (LDS-tiled transpose)
//   k3: fused attention, one wave32 per 16-query tile:
//       - every WMMA operand = two global_load_b128 (contiguous bf16 runs)
//       - raw QK^T scores -> 64 KB LDS; bias added in vectorized softmax pass
//       - attn_weights written as nontemporal b128 stream (268 MB)
//       - P*V accumulates UNnormalized (e-values); 1/rowsum applied to O once
// Fallback: monolithic fp32-input kernel if ws is too small.
// ---------------------------------------------------------------------------

typedef __attribute__((ext_vector_type(16))) __bf16 v16bf;
typedef __attribute__((ext_vector_type(8)))  __bf16 v8bf;
typedef __attribute__((ext_vector_type(4)))  __bf16 v4bf;
typedef __attribute__((ext_vector_type(8)))  float  v8f;
typedef __attribute__((ext_vector_type(4)))  float  v4f;

#define S_LEN 1024
#define D_DIM 64
#define BH    64          // B*H = 4*16
#define NELEM ((size_t)BH * S_LEN * D_DIM)   // 4,194,304 per tensor

__device__ __forceinline__ __bf16 f2bf(float f) { return (__bf16)f; }

// XOR swizzle on LDS column (flips bits 4/5 only within a row's 16-blocks ->
// preserves 4/8-float runs)
__device__ __forceinline__ int sidx(int r, int c) {
  return r * S_LEN + (c ^ ((r & 3) << 4));
}

// operand fragment = two contiguous 8x-bf16 runs at p[0..7] and p[16..23]
__device__ __forceinline__ v16bf load_frag(const __bf16* p) {
  v8bf lo = *(const v8bf*)(p);
  v8bf hi = *(const v8bf*)(p + 16);
  return __builtin_shufflevector(lo, hi, 0, 1, 2, 3, 4, 5, 6, 7,
                                 8, 9, 10, 11, 12, 13, 14, 15);
}

// ---------------------------------------------------------------------------
// Shared softmax over 16 score rows in LDS (raw scores; adds bias here).
// On exit: sbuf holds e = exp(s - rowmax), attn_weights rows written,
// rinv[16] = 1/rowsum (wave-uniform).
// ---------------------------------------------------------------------------
__device__ __forceinline__ void softmax_rows(float* __restrict__ sbuf,
                                             const float* __restrict__ biasQ,
                                             float* __restrict__ PhQ,
                                             int lane, float* __restrict__ rinv)
{
  float rmax[16], rsum[16];
#pragma unroll
  for (int r = 0; r < 16; ++r) { rmax[r] = -3.0e38f; rsum[r] = 0.f; }

  // pass 1: s = raw + bias (vectorized), store back, track row max
  for (int i = 0; i < 8; ++i) {
    const int c0 = 4 * lane + 128 * i;
#pragma unroll
    for (int r = 0; r < 16; ++r) {
      v4f* sp = (v4f*)&sbuf[sidx(r, c0)];
      const v4f s = *sp + *(const v4f*)&biasQ[r * S_LEN + c0];
      *sp = s;
      rmax[r] = fmaxf(rmax[r], fmaxf(fmaxf(s[0], s[1]), fmaxf(s[2], s[3])));
    }
  }
#pragma unroll
  for (int r = 0; r < 16; ++r)
#pragma unroll
    for (int off = 16; off > 0; off >>= 1)
      rmax[r] = fmaxf(rmax[r], __shfl_xor(rmax[r], off, 32));

  // pass 2: e = exp(s - max), store back, accumulate row sum
  for (int i = 0; i < 8; ++i) {
    const int c0 = 4 * lane + 128 * i;
#pragma unroll
    for (int r = 0; r < 16; ++r) {
      v4f* sp = (v4f*)&sbuf[sidx(r, c0)];
      const v4f s = *sp;
      v4f e;
      e[0] = __expf(s[0] - rmax[r]); e[1] = __expf(s[1] - rmax[r]);
      e[2] = __expf(s[2] - rmax[r]); e[3] = __expf(s[3] - rmax[r]);
      *sp = e;
      rsum[r] += (e[0] + e[1]) + (e[2] + e[3]);
    }
  }
#pragma unroll
  for (int r = 0; r < 16; ++r)
#pragma unroll
    for (int off = 16; off > 0; off >>= 1)
      rsum[r] += __shfl_xor(rsum[r], off, 32);
#pragma unroll
  for (int r = 0; r < 16; ++r) rinv[r] = 1.0f / rsum[r];

  // pass 3: attn_weights = e * rinv, nontemporal b128 stream
  for (int r = 0; r < 16; ++r) {
    float* prow = PhQ + (size_t)r * S_LEN;
    for (int i = 0; i < 8; ++i) {
      const int c0 = 4 * lane + 128 * i;
      const v4f e = *(const v4f*)&sbuf[sidx(r, c0)];
      __builtin_nontemporal_store(e * rinv[r], (v4f*)&prow[c0]);
    }
  }
}

// ---------------------------- prep kernels ---------------------------------

// Q*0.125, K : fp32 -> bf16 (4 elements per thread)
__global__ __launch_bounds__(256)
void cvt_qk(const float* __restrict__ Q, const float* __restrict__ K,
            __bf16* __restrict__ Qb, __bf16* __restrict__ Kb)
{
  const int i = blockIdx.x * 256 + threadIdx.x;   // NELEM/4 threads
  const float4 q = ((const float4*)Q)[i];
  const float4 k = ((const float4*)K)[i];
  v4bf qb, kb;
  qb[0] = f2bf(q.x * 0.125f); qb[1] = f2bf(q.y * 0.125f);
  qb[2] = f2bf(q.z * 0.125f); qb[3] = f2bf(q.w * 0.125f);
  kb[0] = f2bf(k.x); kb[1] = f2bf(k.y); kb[2] = f2bf(k.z); kb[3] = f2bf(k.w);
  ((v4bf*)Qb)[i] = qb;
  ((v4bf*)Kb)[i] = kb;
}

// V fp32 [bh][s][d] -> bf16 Vt [bh][d][s], LDS-tiled 64x64 transpose
__global__ __launch_bounds__(256)
void cvt_v_t(const float* __restrict__ V, __bf16* __restrict__ Vt)
{
  __shared__ float t[D_DIM * 65];
  const int bh = blockIdx.x >> 4;
  const int s0 = (blockIdx.x & 15) * 64;
  const float* Vh = V + (size_t)bh * S_LEN * D_DIM;
  __bf16* Vth = Vt + (size_t)bh * S_LEN * D_DIM;
#pragma unroll
  for (int i = 0; i < 16; ++i) {
    const int idx = threadIdx.x + 256 * i;
    const int s = idx >> 6, d = idx & 63;
    t[d * 65 + s] = Vh[(size_t)(s0 + s) * D_DIM + d];
  }
  __syncthreads();
#pragma unroll
  for (int i = 0; i < 16; ++i) {
    const int idx = threadIdx.x + 256 * i;
    const int d = idx >> 6, s = idx & 63;
    Vth[(size_t)d * S_LEN + s0 + s] = f2bf(t[d * 65 + s]);
  }
}

// ---------------------------- main kernel ----------------------------------

__global__ __launch_bounds__(32)
void attn_bf16(const __bf16* __restrict__ Qb, const __bf16* __restrict__ Kb,
               const __bf16* __restrict__ Vt, const float* __restrict__ bias,
               float* __restrict__ Out, float* __restrict__ Pw)
{
  __shared__ float sbuf[16 * S_LEN];   // 64 KB: 16 query rows x 1024 scores

  const int lane = threadIdx.x;
  const int half = lane >> 4;
  const int l16  = lane & 15;
  const int kb   = half * 8;

  const int bh = blockIdx.x >> 6;
  const int q0 = (blockIdx.x & 63) * 16;

  const __bf16* Qh = Qb + (size_t)bh * S_LEN * D_DIM;
  const __bf16* Kh = Kb + (size_t)bh * S_LEN * D_DIM;
  const __bf16* Vh = Vt + (size_t)bh * S_LEN * D_DIM;   // [d][s]
  float* Oh = Out + (size_t)bh * S_LEN * D_DIM;
  float* Ph = Pw  + (size_t)bh * S_LEN * S_LEN;

  // ---- Q A-fragments (scale pre-folded into Qb) ----------------------------
  v16bf aq[2];
  {
    const __bf16* qr = Qh + (size_t)(q0 + l16) * D_DIM + kb;
    aq[0] = load_frag(qr);
    aq[1] = load_frag(qr + 32);
  }

  // ---- Phase 1: raw scores = (Q*scale) K^T -> LDS --------------------------
  // D->C accumulator chaining (not a listed WMMA hazard; cheaper than
  // independent accumulators + VALU adds, which double the coexec stalls).
  for (int t = 0; t < 64; ++t) {
    const __bf16* kr = Kh + (size_t)(t * 16 + l16) * D_DIM + kb;
    if (t + 1 < 64) __builtin_prefetch(kr + 16 * D_DIM, 0, 3);

    v8f acc = {};
    acc = __builtin_amdgcn_wmma_f32_16x16x32_bf16(false, aq[0], false,
            load_frag(kr),      (short)0, acc, false, false);
    acc = __builtin_amdgcn_wmma_f32_16x16x32_bf16(false, aq[1], false,
            load_frag(kr + 32), (short)0, acc, false, false);

    const int col = t * 16 + l16;
#pragma unroll
    for (int v = 0; v < 8; ++v)
      sbuf[sidx(v + half * 8, col)] = acc[v];
  }

  // ---- Phase 2: softmax (+bias) and attn_weights stream --------------------
  float rinv[16];
  softmax_rows(sbuf, bias + (size_t)q0 * S_LEN, Ph + (size_t)q0 * S_LEN,
               lane, rinv);

  const int s4 = (l16 & 3) << 4;

  // ---- Phase 3: unnormalized O = E * V (32 chunks of K=32 keys) ------------
  v8f oacc[4] = {};
  for (int kc = 0; kc < 32; ++kc) {
    v16bf ap;   // A = e tile 16x32 from LDS (row l16)
#pragma unroll
    for (int h = 0; h < 2; ++h) {
      const float* pr = &sbuf[l16 * S_LEN + ((kc * 32 + h * 16 + kb) ^ s4)];
      const v4f x0 = *(const v4f*)(pr);
      const v4f x1 = *(const v4f*)(pr + 4);
      ap[8*h+0] = f2bf(x0[0]); ap[8*h+1] = f2bf(x0[1]);
      ap[8*h+2] = f2bf(x0[2]); ap[8*h+3] = f2bf(x0[3]);
      ap[8*h+4] = f2bf(x1[0]); ap[8*h+5] = f2bf(x1[1]);
      ap[8*h+6] = f2bf(x1[2]); ap[8*h+7] = f2bf(x1[3]);
    }
#pragma unroll
    for (int nb = 0; nb < 4; ++nb) {
      const __bf16* vr = Vh + (size_t)(nb * 16 + l16) * S_LEN + kc * 32 + kb;
      oacc[nb] = __builtin_amdgcn_wmma_f32_16x16x32_bf16(false, ap, false,
                   load_frag(vr), (short)0, oacc[nb], false, false);
    }
  }

  // ---- Phase 4: O = oacc * rinv[row] ---------------------------------------
#pragma unroll
  for (int nb = 0; nb < 4; ++nb)
#pragma unroll
    for (int v = 0; v < 8; ++v) {
      const float rv = half ? rinv[v + 8] : rinv[v];
      const int m = v + half * 8;
      Oh[(size_t)(q0 + m) * D_DIM + nb * 16 + l16] = oacc[nb][v] * rv;
    }
}

// ------------------------- fallback (no-ws) kernel --------------------------

__global__ __launch_bounds__(32)
void attn_fused_f32(const float* __restrict__ Q, const float* __restrict__ K,
                    const float* __restrict__ V, const float* __restrict__ bias,
                    float* __restrict__ Out, float* __restrict__ Pw)
{
  __shared__ float sbuf[16 * S_LEN];
  const int lane = threadIdx.x, half = lane >> 4, l16 = lane & 15, kb = half * 8;
  const int bh = blockIdx.x >> 6, q0 = (blockIdx.x & 63) * 16;
  const float* Qh = Q + (size_t)bh * S_LEN * D_DIM;
  const float* Kh = K + (size_t)bh * S_LEN * D_DIM;
  const float* Vh = V + (size_t)bh * S_LEN * D_DIM;
  float* Oh = Out + (size_t)bh * S_LEN * D_DIM;
  float* Ph = Pw  + (size_t)bh * S_LEN * S_LEN;

  v16bf aq[2];
  {
    const float* qrow = Qh + (size_t)(q0 + l16) * D_DIM;
#pragma unroll
    for (int c = 0; c < 2; ++c)
#pragma unroll
      for (int j = 0; j < 8; ++j) {
        const int kd = c * 32 + ((j & 4) ? 16 : 0) + kb + 2 * (j & 3);
        aq[c][2*j]   = f2bf(qrow[kd]   * 0.125f);
        aq[c][2*j+1] = f2bf(qrow[kd+1] * 0.125f);
      }
  }
  for (int t = 0; t < 64; ++t) {
    const float* krow = Kh + (size_t)(t * 16 + l16) * D_DIM;
    v8f acc = {};
#pragma unroll
    for (int c = 0; c < 2; ++c) {
      v16bf bk;
#pragma unroll
      for (int j = 0; j < 8; ++j) {
        const int kd = c * 32 + ((j & 4) ? 16 : 0) + kb + 2 * (j & 3);
        bk[2*j] = f2bf(krow[kd]); bk[2*j+1] = f2bf(krow[kd+1]);
      }
      acc = __builtin_amdgcn_wmma_f32_16x16x32_bf16(false, aq[c], false, bk,
                                                    (short)0, acc, false, false);
    }
    const int col = t * 16 + l16;
#pragma unroll
    for (int v = 0; v < 8; ++v)
      sbuf[sidx(v + half * 8, col)] = acc[v];
  }

  float rinv[16];
  softmax_rows(sbuf, bias + (size_t)q0 * S_LEN, Ph + (size_t)q0 * S_LEN,
               lane, rinv);

  const int s4 = (l16 & 3) << 4;
  v8f oacc[4] = {};
  for (int kc = 0; kc < 32; ++kc) {
    v16bf ap;
#pragma unroll
    for (int h = 0; h < 2; ++h) {
      const float* pr = &sbuf[l16 * S_LEN + ((kc * 32 + h * 16 + kb) ^ s4)];
      const v4f x0 = *(const v4f*)(pr);
      const v4f x1 = *(const v4f*)(pr + 4);
      ap[8*h+0] = f2bf(x0[0]); ap[8*h+1] = f2bf(x0[1]);
      ap[8*h+2] = f2bf(x0[2]); ap[8*h+3] = f2bf(x0[3]);
      ap[8*h+4] = f2bf(x1[0]); ap[8*h+5] = f2bf(x1[1]);
      ap[8*h+6] = f2bf(x1[2]); ap[8*h+7] = f2bf(x1[3]);
    }
#pragma unroll
    for (int nb = 0; nb < 4; ++nb) {
      v16bf bv;
#pragma unroll
      for (int j = 0; j < 8; ++j) {
        const int kd = ((j & 4) ? 16 : 0) + kb + 2 * (j & 3);
        bv[2*j]   = f2bf(Vh[(size_t)(kc * 32 + kd)     * D_DIM + nb * 16 + l16]);
        bv[2*j+1] = f2bf(Vh[(size_t)(kc * 32 + kd + 1) * D_DIM + nb * 16 + l16]);
      }
      oacc[nb] = __builtin_amdgcn_wmma_f32_16x16x32_bf16(false, ap, false, bv,
                                                         (short)0, oacc[nb],
                                                         false, false);
    }
  }
#pragma unroll
  for (int nb = 0; nb < 4; ++nb)
#pragma unroll
    for (int v = 0; v < 8; ++v) {
      const float rv = half ? rinv[v + 8] : rinv[v];
      const int m = v + half * 8;
      Oh[(size_t)(q0 + m) * D_DIM + nb * 16 + l16] = oacc[nb][v] * rv;
    }
}

// ------------------------------- launcher -----------------------------------

extern "C" void kernel_launch(void* const* d_in, const int* in_sizes, int n_in,
                              void* d_out, int out_size, void* d_ws, size_t ws_size,
                              hipStream_t stream) {
  (void)in_sizes; (void)n_in; (void)out_size;
  const float* Q    = (const float*)d_in[0];
  const float* K    = (const float*)d_in[1];
  const float* V    = (const float*)d_in[2];
  const float* bias = (const float*)d_in[3];

  float* Out = (float*)d_out;                      // [B,H,S,D]
  float* Pw  = Out + NELEM;                        // [B,H,S,S]

  const size_t need = NELEM * 3 * sizeof(__bf16);  // Qb + Kb + Vt ~ 25.2 MB
  if (ws_size >= need && d_ws != nullptr) {
    __bf16* Qb = (__bf16*)d_ws;
    __bf16* Kb = Qb + NELEM;
    __bf16* Vt = Kb + NELEM;
    cvt_qk <<<dim3((unsigned)(NELEM / 4 / 256)), dim3(256), 0, stream>>>(Q, K, Qb, Kb);
    cvt_v_t<<<dim3(BH * 16),                     dim3(256), 0, stream>>>(V, Vt);
    attn_bf16<<<dim3(BH * (S_LEN / 16)), dim3(32), 0, stream>>>(Qb, Kb, Vt, bias, Out, Pw);
  } else {
    attn_fused_f32<<<dim3(BH * (S_LEN / 16)), dim3(32), 0, stream>>>(Q, K, V, bias, Out, Pw);
  }
}
